// ArgmaxIOU_69114613729374
// MI455X (gfx1250) — 8-bit matrix operand order repaired and checked by reading, compiled, not run on hardware
//
#include <hip/hip_runtime.h>
#include <hip/hip_bf16.h>

#define C_CLS 21
#define B_SZ 8
#define HW (512 * 512)
#define TILE_W 128
#define TILES_PER_BLOCK 8
#define THREADS 128

typedef unsigned int u32x4 __attribute__((ext_vector_type(4)));
typedef int i32x4 __attribute__((ext_vector_type(4)));
typedef int i32x8 __attribute__((ext_vector_type(8)));

__device__ __forceinline__ void tdm_load_tile(const float* gptr, unsigned lds_off) {
    unsigned long long ga = (unsigned long long)(uintptr_t)gptr;

    u32x4 g0;
    g0[0] = 1u;
    g0[1] = lds_off;
    g0[2] = (unsigned)(ga & 0xFFFFFFFFull);
    g0[3] = (unsigned)((ga >> 32) & 0x1FFFFFFull)
            | (2u << 30);

    i32x8 g1;
    g1[0] = (int)(2u << 16);
    g1[1] = (int)((TILE_W & 0xFFFFu) << 16);
    g1[2] = (int)(((TILE_W >> 16) & 0xFFFFu)
            | ((C_CLS & 0xFFFFu) << 16));
    g1[3] = (int)(((C_CLS >> 16) & 0xFFFFu)
            | ((TILE_W & 0xFFFFu) << 16));
    g1[4] = (int)(C_CLS & 0xFFFFu);
    g1[5] = (int)HW;
    g1[6] = 0;
    g1[7] = 0;

    i32x4 z4 = {0, 0, 0, 0};
#if defined(__clang_major__) && (__clang_major__ >= 23)
    i32x8 z8 = {0, 0, 0, 0, 0, 0, 0, 0};
    __builtin_amdgcn_tensor_load_to_lds(g0, g1, z4, z4, z8, 0);
#else
    __builtin_amdgcn_tensor_load_to_lds(g0, g1, z4, z4, 0);
#endif
}

__device__ __forceinline__ unsigned eq_bytes(unsigned d, unsigned mrep) {
    unsigned x = d ^ mrep;
    unsigned s = (x | 0x80808080u) - 0x01010101u;
    return (~s & 0x80808080u) >> 7;
}

__global__ void __launch_bounds__(256) zero_kernel(unsigned* p, int n) {
    int i = blockIdx.x * blockDim.x + threadIdx.x;
    if (i < n) p[i] = 0u;
}

__global__ void __launch_bounds__(THREADS) conf_kernel(const float* __restrict__ pred,
                                                       const float* __restrict__ tgt,
                                                       unsigned* __restrict__ gconf) {
    __shared__ float sp[2][C_CLS * TILE_W];
    __shared__ float st[2][C_CLS * TILE_W];
    __shared__ unsigned s_ytw[4][16];
    __shared__ unsigned s_ypw[4][16];
    __shared__ unsigned sconf[C_CLS * C_CLS];

    const int tid = threadIdx.x;
    const int wave = tid >> 5;
    const unsigned lane = (unsigned)(tid & 31);
    const unsigned half = lane >> 4;

    const int tilesPerB = HW / TILE_W;
    const int gtile0 = blockIdx.x * TILES_PER_BLOCK;
    const int b = gtile0 / tilesPerB;
    const long long baseB = (long long)b * C_CLS * HW;

    for (int i = tid; i < C_CLS * C_CLS; i += THREADS) sconf[i] = 0u;

    const unsigned sp_base = (unsigned)(uintptr_t)(&sp[0][0]);
    const unsigned st_base = (unsigned)(uintptr_t)(&st[0][0]);
    const unsigned bufBytes = C_CLS * TILE_W * 4u;

    i32x8 c00 = {0, 0, 0, 0, 0, 0, 0, 0};
    i32x8 c01 = {0, 0, 0, 0, 0, 0, 0, 0};
    i32x8 c10 = {0, 0, 0, 0, 0, 0, 0, 0};
    i32x8 c11 = {0, 0, 0, 0, 0, 0, 0, 0};
    const unsigned mrep0 = (lane & 15u) * 0x01010101u;
    const unsigned mrep1 = mrep0 + 0x10101010u;

    if (tid < 32) {
        const long long off0 = baseB + (long long)(gtile0 % tilesPerB) * TILE_W;
        tdm_load_tile(pred + off0, sp_base);
        tdm_load_tile(tgt + off0, st_base);
    }

    for (int i = 0; i < TILES_PER_BLOCK; ++i) {
        if (tid < 32) {
            if (i + 1 < TILES_PER_BLOCK) {
                const int nb = (i + 1) & 1;
                const long long offn =
                    baseB + (long long)((gtile0 + i + 1) % tilesPerB) * TILE_W;
                tdm_load_tile(pred + offn, sp_base + nb * bufBytes);
                tdm_load_tile(tgt + offn, st_base + nb * bufBytes);
                __builtin_amdgcn_s_wait_tensorcnt(2);
            } else {
                __builtin_amdgcn_s_wait_tensorcnt(0);
            }
        }
        __syncthreads();

        const float* P = sp[i & 1];
        const float* T = st[i & 1];
        float bp = P[tid];
        float bt = T[tid];
        int ip = 0, it = 0;
#pragma unroll
        for (int c = 1; c < C_CLS; ++c) {
            float v = P[c * TILE_W + tid];
            if (v > bp) { bp = v; ip = c; }
            float w = T[c * TILE_W + tid];
            if (w > bt) { bt = w; it = c; }
        }
        const int slot = ((i & 1) << 5) + (int)lane;
        ((unsigned char*)&s_ytw[wave][0])[slot] = (unsigned char)it;
        ((unsigned char*)&s_ypw[wave][0])[slot] = (unsigned char)ip;
        __syncthreads();

        if (i & 1) {
            const unsigned* ytd = &s_ytw[wave][0];
            const unsigned* ypd = &s_ypw[wave][0];
            i32x8 a0, a1, b0, b1;
#pragma unroll
            for (int v = 0; v < 8; ++v) {
                unsigned da =
                    ytd[(v & 1) + (((v >> 1) & 1) << 2) + ((v >> 2) << 3) + (half << 1)];
                a0[v] = (int)eq_bytes(da, mrep0);
                a1[v] = (int)eq_bytes(da, mrep1);
                unsigned db = ypd[(v & 1) + (((v >> 1) & 1) << 2) + ((v >> 2) << 3) + (half << 1)];
                b0[v] = (int)eq_bytes(db, mrep0);
                b1[v] = (int)eq_bytes(db, mrep1);
            }
            c00 = __builtin_amdgcn_wmma_i32_16x16x64_iu8(false, a0, false, b0, c00,
                                                         false, false);
            c01 = __builtin_amdgcn_wmma_i32_16x16x64_iu8(false, a0, false, b1, c01,
                                                         false, false);
            c10 = __builtin_amdgcn_wmma_i32_16x16x64_iu8(false, a1, false, b0, c10,
                                                         false, false);
            c11 = __builtin_amdgcn_wmma_i32_16x16x64_iu8(false, a1, false, b1, c11,
                                                         false, false);
        }
    }

    {
        const int n0 = (int)(lane & 15u);
        const int mbase = (int)(half << 3);
#pragma unroll
        for (int r = 0; r < 8; ++r) {
            const int m0 = mbase + r;
            if (c00[r]) atomicAdd(&sconf[m0 * C_CLS + n0], (unsigned)c00[r]);
            if (n0 + 16 < C_CLS && c01[r])
                atomicAdd(&sconf[m0 * C_CLS + n0 + 16], (unsigned)c01[r]);
            if (m0 + 16 < C_CLS && c10[r])
                atomicAdd(&sconf[(m0 + 16) * C_CLS + n0], (unsigned)c10[r]);
            if (m0 + 16 < C_CLS && n0 + 16 < C_CLS && c11[r])
                atomicAdd(&sconf[(m0 + 16) * C_CLS + n0 + 16], (unsigned)c11[r]);
        }
    }
    __syncthreads();

    for (int i = tid; i < C_CLS * C_CLS; i += THREADS) {
        unsigned cnt = sconf[i];
        if (cnt) atomicAdd(&gconf[b * C_CLS * C_CLS + i], cnt);
    }
}

__global__ void finalize_kernel(const unsigned* __restrict__ gconf,
                                float* __restrict__ out) {
    if (threadIdx.x == 0 && blockIdx.x == 0) {
        float acc = 0.f;
        for (int b = 0; b < B_SZ; ++b) {
            const unsigned* m = gconf + b * C_CLS * C_CLS;
            float iousum = 0.f, nvalid = 0.f;
            for (int c = 0; c < C_CLS; ++c) {
                float tp = (float)m[c * C_CLS + c];
                float row = 0.f, col = 0.f;
                for (int k = 0; k < C_CLS; ++k) {
                    row += (float)m[c * C_CLS + k];
                    col += (float)m[k * C_CLS + c];
                }
                float fn = row - tp;
                float fp = col - tp;
                if (tp > 0.f) {
                    iousum += tp / (tp + fn + fp);
                    nvalid += 1.f;
                }
            }
            float denomv = nvalid > 1.f ? nvalid : 1.f;
            acc += iousum / denomv;
        }
        *out = acc / (float)B_SZ;
    }
}

extern "C" void kernel_launch(void* const* d_in, const int* in_sizes, int n_in,
                              void* d_out, int out_size, void* d_ws, size_t ws_size,
                              hipStream_t stream) {
    const float* pred = (const float*)d_in[0];
    const float* tgt = (const float*)d_in[1];
    unsigned* gconf = (unsigned*)d_ws;

    const int nconf = B_SZ * C_CLS * C_CLS;
    zero_kernel<<<(nconf + 255) / 256, 256, 0, stream>>>(gconf, nconf);

    const int totalTiles = B_SZ * (HW / TILE_W);
    const int blocks = totalTiles / TILES_PER_BLOCK;
    conf_kernel<<<blocks, THREADS, 0, stream>>>(pred, tgt, gconf);

    finalize_kernel<<<1, 32, 0, stream>>>(gconf, (float*)d_out);
}
